// TransformerBlock_66056597012611
// MI455X (gfx1250) — matrix operand order repaired and checked by reading, not compiled, not run on hardware
//
#include <hip/hip_runtime.h>

// ---------------------------------------------------------------------------
// Transformer block for gfx1250 (MI455X), bf16 WMMA everywhere.
// B=2, T=2048, H=2048, NH=16, HD=128
// ---------------------------------------------------------------------------

#define TB 256

constexpr int Bc  = 2;
constexpr int Tc  = 2048;
constexpr int Hc  = 2048;
constexpr int NHc = 16;
constexpr int HDc = 128;
constexpr int H4c = 8192;
constexpr int BTc = Bc * Tc;

typedef __attribute__((ext_vector_type(16))) __bf16 v16bf;
typedef __attribute__((ext_vector_type(8)))  float  v8f;

union FragU { uint4 q[2]; v16bf v; };

// Async global->LDS copies (CDNA5 GLOBAL_LOAD_ASYNC_TO_LDS_B128, ASYNCcnt).
// Guarded so the build falls back to sync staging if the builtin is absent.
#if defined(__gfx1250__) && __has_builtin(__builtin_amdgcn_global_load_async_to_lds_b128) && \
    __has_builtin(__builtin_amdgcn_s_wait_asynccnt)
#define HAVE_ASYNC_LDS 1
#endif

#if defined(HAVE_ASYNC_LDS)
// Parameter type per hipcc diagnostic: 'int __attribute__((vector_size(16))) __device__ *'
typedef int async_i4 __attribute__((vector_size(16)));
static __device__ __forceinline__ void async_copy_b128(const void* g, void* l) {
  __builtin_amdgcn_global_load_async_to_lds_b128(
      (__attribute__((address_space(1))) async_i4*)(g),
      (__attribute__((address_space(3))) async_i4*)(l), 0, 0);
}
#endif

static __device__ __forceinline__ unsigned short f2bf_raw(float f) {
  unsigned int u = __float_as_uint(f);
  u += 0x7FFFu + ((u >> 16) & 1u);           // round-to-nearest-even
  return (unsigned short)(u >> 16);
}

static __device__ __forceinline__ v8f wmma_bf16(v16bf a, v16bf b, v8f c) {
  // D = A(16x32 bf16) * B(32x16 bf16) + C(16x16 f32)
  return __builtin_amdgcn_wmma_f32_16x16x32_bf16(false, a, false, b,
                                                 (short)0, c, false, false);
}

// A-matrix fragment: lane<16 holds K {k0..k0+7, k0+16..k0+23} of row (lane&15);
// lane>=16 holds K {k0+8..k0+15, k0+24..k0+31}. Tile is [row][k], rs halves/row.
static __device__ __forceinline__ v16bf fragA16(const unsigned short* t,
                                                int row0, int rs, int k0) {
  const int lane = threadIdx.x & 31;
  const int off = (row0 + (lane & 15)) * rs + k0 + ((lane >> 4) << 3);
  FragU u;
  u.q[0] = *reinterpret_cast<const uint4*>(t + off);
  u.q[1] = *reinterpret_cast<const uint4*>(t + off + 16);
  return u.v;
}

// B-matrix fragment: lane<16 holds K {k0..k0+7, k0+16..k0+23} of column (lane&15);
// lane>=16 holds K {k0+8..k0+15, k0+24..k0+31}. Tile is [col][k] (i.e. B transposed), rs halves/row.
static __device__ __forceinline__ v16bf fragB16(const unsigned short* t,
                                                int col0, int rs, int k0) {
  const int lane = threadIdx.x & 31;
  const int off = (col0 + (lane & 15)) * rs + k0 + ((lane >> 4) << 3);
  FragU u;
  u.q[0] = *reinterpret_cast<const uint4*>(t + off);
  u.q[1] = *reinterpret_cast<const uint4*>(t + off + 16);
  return u.v;
}

static __device__ __forceinline__ float gelu_tanh_f(float x) {
  const float c = 0.7978845608028654f;       // sqrt(2/pi)
  const float x3 = x * x * x;
  return 0.5f * x * (1.0f + tanhf(c * (x + 0.044715f * x3)));
}

// ---------------------------------------------------------------------------
// Cast fp32 [K][N] weight -> bf16 [N][K] (transposed), LDS-tiled 64x64 so both
// global reads (fp32 along N) and writes (bf16 along K) are coalesced.
// ---------------------------------------------------------------------------
__global__ __launch_bounds__(TB)
void cast_transpose_kernel(const float* __restrict__ in,
                           unsigned short* __restrict__ out, int K, int N) {
  __shared__ unsigned short tile[64][65];    // pad to dodge bank conflicts
  const int tid = threadIdx.x;
  const int kb = blockIdx.y * 64;
  const int nb = blockIdx.x * 64;
#pragma unroll
  for (int i = 0; i < 16; ++i) {
    const int idx = i * TB + tid;            // 0..4095
    const int r = idx >> 6;                  // k-local
    const int c = idx & 63;                  // n-local (coalesced read)
    tile[c][r] = f2bf_raw(in[(size_t)(kb + r) * N + nb + c]);
  }
  __syncthreads();
#pragma unroll
  for (int i = 0; i < 16; ++i) {
    const int idx = i * TB + tid;
    const int r = idx >> 6;                  // n-local
    const int c = idx & 63;                  // k-local (coalesced write)
    out[(size_t)(nb + r) * K + kb + c] = tile[r][c];
  }
}

// ---------------------------------------------------------------------------
// LayerNorm over last dim (H=2048), fp32 in -> bf16 out. One block per row.
// ---------------------------------------------------------------------------
__global__ __launch_bounds__(TB)
void layernorm_bf16_kernel(const float* __restrict__ x,
                           const float* __restrict__ sc,
                           const float* __restrict__ sh,
                           unsigned short* __restrict__ out) {
  __shared__ float red[TB];
  const int row = blockIdx.x;
  const int tid = threadIdx.x;
  const float* xr = x + (size_t)row * Hc;
  float lv[8];
  float s = 0.f, ss = 0.f;
#pragma unroll
  for (int i = 0; i < 8; ++i) {
    const float v0 = xr[tid + i * TB];
    lv[i] = v0; s += v0; ss += v0 * v0;
  }
  red[tid] = s; __syncthreads();
  for (int st = TB / 2; st > 0; st >>= 1) {
    if (tid < st) red[tid] += red[tid + st];
    __syncthreads();
  }
  const float mean = red[0] * (1.0f / Hc);
  __syncthreads();
  red[tid] = ss; __syncthreads();
  for (int st = TB / 2; st > 0; st >>= 1) {
    if (tid < st) red[tid] += red[tid + st];
    __syncthreads();
  }
  const float var = red[0] * (1.0f / Hc) - mean * mean;
  const float rstd = rsqrtf(var + 1e-5f);
  unsigned short* orow = out + (size_t)row * Hc;
#pragma unroll
  for (int i = 0; i < 8; ++i) {
    const int c = tid + i * TB;
    orow[c] = f2bf_raw(sc[c] * (lv[i] - mean) * rstd + sh[c]);
  }
}

// ---------------------------------------------------------------------------
// bf16 WMMA GEMM: C[M,N] = A[M,K] @ B[K,N], B given transposed as Bt[N,K].
// Block tile 128x128, BK=32, 8 waves (4x2), each wave 32x64 = 8 WMMA tiles.
// When available, tiles stream in via async global->LDS copies with double
// buffering (ASYNCcnt); otherwise sync global_load + ds_store staging.
// ---------------------------------------------------------------------------
enum { EPI_BF16 = 0, EPI_RES_F32 = 1, EPI_BIAS_GELU_BF16 = 2, EPI_BIAS_RES_F32 = 3 };

template <int EPI>
__global__ __launch_bounds__(TB)
void gemm_bf16_kernel(const unsigned short* __restrict__ A,
                      const unsigned short* __restrict__ Bt,
                      const float* __restrict__ bias,
                      const float* __restrict__ res,
                      void* __restrict__ outp,
                      int M, int N, int K) {
  const int tid  = threadIdx.x;
  const int lane = tid & 31;
  const int wid  = tid >> 5;
  const int wm   = wid >> 1;                // 0..3 (32-row slab)
  const int wn   = wid & 1;                 // 0..1 (64-col slab)
  const int m0   = blockIdx.y * 128;
  const int n0   = blockIdx.x * 128;

  const v8f vzero = {0.f, 0.f, 0.f, 0.f, 0.f, 0.f, 0.f, 0.f};
  v8f acc[2][4];
#pragma unroll
  for (int s = 0; s < 2; ++s)
#pragma unroll
    for (int j = 0; j < 4; ++j) acc[s][j] = vzero;

#if defined(HAVE_ASYNC_LDS)
  __shared__ unsigned short As[2][128 * 32];  // [m][k] double-buffered
  __shared__ unsigned short Bs[2][128 * 32];  // [n][k] double-buffered

  auto stage_async = [&](int buf, int kt) {
#pragma unroll
    for (int i = 0; i < 2; ++i) {
      const int idx = tid * 2 + i;          // 0..511
      const int row = idx >> 2;             // 0..127
      const int c8  = (idx & 3) << 3;       // 0,8,16,24 (halves)
      async_copy_b128(A + (size_t)(m0 + row) * K + kt + c8, &As[buf][row * 32 + c8]);
      async_copy_b128(Bt + (size_t)(n0 + row) * K + kt + c8, &Bs[buf][row * 32 + c8]);
    }
  };

  stage_async(0, 0);
  __builtin_amdgcn_s_wait_asynccnt(0);
  __syncthreads();
  int cur = 0;
  for (int kt = 0; kt < K; kt += 32) {
    if (kt + 32 < K) stage_async(cur ^ 1, kt + 32);  // overlap with WMMA below
    const v16bf af0 = fragA16(As[cur], wm * 32, 32, 0);
    const v16bf af1 = fragA16(As[cur], wm * 32 + 16, 32, 0);
#pragma unroll
    for (int j = 0; j < 4; ++j) {
      const v16bf bfr = fragB16(Bs[cur], wn * 64 + j * 16, 32, 0);
      acc[0][j] = wmma_bf16(af0, bfr, acc[0][j]);
      acc[1][j] = wmma_bf16(af1, bfr, acc[1][j]);
    }
    __builtin_amdgcn_s_wait_asynccnt(0);    // next tile landed in LDS
    __syncthreads();                        // all waves done reading cur
    cur ^= 1;
  }
#else
  __shared__ unsigned short As[128 * 32];   // [m][k]
  __shared__ unsigned short Bs[128 * 32];   // [n][k]
  for (int kt = 0; kt < K; kt += 32) {
    __syncthreads();
#pragma unroll
    for (int i = 0; i < 2; ++i) {
      const int idx = tid * 2 + i;          // 0..511
      const int row = idx >> 2;             // 0..127
      const int c8  = (idx & 3) << 3;       // 0,8,16,24 (halves)
      *reinterpret_cast<uint4*>(&As[row * 32 + c8]) =
          *reinterpret_cast<const uint4*>(A + (size_t)(m0 + row) * K + kt + c8);
      *reinterpret_cast<uint4*>(&Bs[row * 32 + c8]) =
          *reinterpret_cast<const uint4*>(Bt + (size_t)(n0 + row) * K + kt + c8);
    }
    if (kt + 32 < K) {
      const int row = (tid * 2) >> 2;
      __builtin_prefetch(A + (size_t)(m0 + row) * K + kt + 32, 0, 0);
      __builtin_prefetch(Bt + (size_t)(n0 + row) * K + kt + 32, 0, 0);
    }
    __syncthreads();
    const v16bf af0 = fragA16(As, wm * 32, 32, 0);
    const v16bf af1 = fragA16(As, wm * 32 + 16, 32, 0);
#pragma unroll
    for (int j = 0; j < 4; ++j) {
      const v16bf bfr = fragB16(Bs, wn * 64 + j * 16, 32, 0);
      acc[0][j] = wmma_bf16(af0, bfr, acc[0][j]);
      acc[1][j] = wmma_bf16(af1, bfr, acc[1][j]);
    }
  }
#endif

  const int nloc = lane & 15;
  const int rhi  = (lane >> 4) << 3;
#pragma unroll
  for (int s = 0; s < 2; ++s)
#pragma unroll
    for (int j = 0; j < 4; ++j)
#pragma unroll
      for (int r = 0; r < 8; ++r) {
        const int row = m0 + wm * 32 + s * 16 + r + rhi;
        const int col = n0 + wn * 64 + j * 16 + nloc;
        float vv = acc[s][j][r];
        if (EPI == EPI_BIAS_GELU_BF16 || EPI == EPI_BIAS_RES_F32) vv += bias[col];
        if (EPI == EPI_BIAS_GELU_BF16) vv = gelu_tanh_f(vv);
        if (EPI == EPI_RES_F32 || EPI == EPI_BIAS_RES_F32)
          vv += res[(size_t)row * N + col];
        if (EPI == EPI_BF16 || EPI == EPI_BIAS_GELU_BF16)
          reinterpret_cast<unsigned short*>(outp)[(size_t)row * N + col] = f2bf_raw(vv);
        else
          reinterpret_cast<float*>(outp)[(size_t)row * N + col] = vv;
      }
}

// ---------------------------------------------------------------------------
// Flash attention: one block per (64-query tile, head, batch). Causal.
// Q/K/V bf16 tiles in LDS; S block in LDS fp32; online softmax; WMMA for
// QK^T and P@V. LDS = 64KB exactly (Pt/red/cI/lI alias the S buffer).
// ---------------------------------------------------------------------------
__global__ __launch_bounds__(TB)
void flash_attn_kernel(const unsigned short* __restrict__ qbuf,
                       const unsigned short* __restrict__ kbuf,
                       const unsigned short* __restrict__ vbuf,
                       unsigned short* __restrict__ obuf) {
  extern __shared__ unsigned char smem_raw[];
  unsigned short* Qt = reinterpret_cast<unsigned short*>(smem_raw); // [64][128]
  unsigned short* Kt = Qt + 64 * 128;                               // [64][128]
  unsigned short* Vt = Kt + 64 * 128;                               // [128][64] (V^T)
  float* Sb = reinterpret_cast<float*>(Vt + 128 * 64);              // [64][64] f32
  unsigned short* Pt = reinterpret_cast<unsigned short*>(Sb);       // alias: [64][64] bf16
  float* red = Sb + 2048;                                           // alias: 256 f32
  float* cI  = Sb + 2304;                                           // alias: 64 f32
  float* lIs = Sb + 2368;                                           // alias: 64 f32

  const int tid  = threadIdx.x;
  const int lane = tid & 31;
  const int wid  = tid >> 5;
  const int nloc = lane & 15;
  const int rhi  = (lane >> 4) << 3;

  const int qblk = blockIdx.x;
  const int q0   = qblk * 64;
  const int h    = blockIdx.y;
  const int bidx = blockIdx.z;
  const size_t rowbase = (size_t)bidx * Tc + q0;
  const size_t hoff    = (size_t)h * HDc;

  // Load Q tile (64x128 bf16) row-major.
#pragma unroll
  for (int i = 0; i < 4; ++i) {
    const int idx = tid * 4 + i;            // 0..1023 uint4s
    const int row = idx >> 4;
    const int c8  = (idx & 15) << 3;
    *reinterpret_cast<uint4*>(&Qt[row * 128 + c8]) =
        *reinterpret_cast<const uint4*>(qbuf + (rowbase + row) * Hc + hoff + c8);
  }

  const v8f vzero = {0.f, 0.f, 0.f, 0.f, 0.f, 0.f, 0.f, 0.f};
  v8f oacc[4];
#pragma unroll
  for (int j = 0; j < 4; ++j) oacc[j] = vzero;

  float m_priv = -3.0e38f;                  // running row max (scaled domain)
  float l_priv = 0.0f;                      // running row sum
  const int rr   = tid >> 2;                // softmax row 0..63
  const int cseg = (tid & 3) << 4;          // 16-col segment
  const float sm_scale = 0.08838834764831845f;  // 1/sqrt(128)

  for (int kb = 0; kb <= qblk; ++kb) {
    __syncthreads();                        // (A) prior reads of Kt/Vt/Pt done
    const size_t kbase = (size_t)bidx * Tc + (size_t)kb * 64;
#pragma unroll
    for (int i = 0; i < 4; ++i) {           // K tile [64][128]
      const int idx = tid * 4 + i;
      const int row = idx >> 4;
      const int c8  = (idx & 15) << 3;
      *reinterpret_cast<uint4*>(&Kt[row * 128 + c8]) =
          *reinterpret_cast<const uint4*>(kbuf + (kbase + row) * Hc + hoff + c8);
    }
#pragma unroll
    for (int i = 0; i < 16; ++i) {          // V tile transposed -> [128][64]
      const int idx = tid * 16 + i;         // 0..4095 uints
      const int row = idx >> 6;
      const int col = (idx & 63) << 1;
      const unsigned int w =
          *reinterpret_cast<const unsigned int*>(vbuf + (kbase + row) * Hc + hoff + col);
      Vt[col * 64 + row]       = (unsigned short)(w & 0xFFFFu);
      Vt[(col + 1) * 64 + row] = (unsigned short)(w >> 16);
    }
    __syncthreads();                        // (B) tiles ready

    // S(64x64) = Q @ K^T : wave (wid&3) = m-tile, (wid>>2) = 32-col half.
    const int swm = wid & 3;
    const int snb = (wid >> 2) << 5;
    v8f sacc0 = vzero, sacc1 = vzero;
#pragma unroll
    for (int k0 = 0; k0 < 128; k0 += 32) {
      const v16bf aq = fragA16(Qt, swm * 16, 128, k0);
      sacc0 = wmma_bf16(aq, fragB16(Kt, snb, 128, k0), sacc0);
      sacc1 = wmma_bf16(aq, fragB16(Kt, snb + 16, 128, k0), sacc1);
    }
#pragma unroll
    for (int r = 0; r < 8; ++r) {
      Sb[(swm * 16 + r + rhi) * 64 + snb + nloc]      = sacc0[r] * sm_scale;
      Sb[(swm * 16 + r + rhi) * 64 + snb + 16 + nloc] = sacc1[r] * sm_scale;
    }
    __syncthreads();                        // (C) S ready

    // Online softmax: 4 threads per row, 16 cols each.
    float vals[16];
    float lmax = -3.0e38f;
    const int qpos = q0 + rr;
#pragma unroll
    for (int c = 0; c < 16; ++c) {
      const int kpos = kb * 64 + cseg + c;
      float sv = Sb[rr * 64 + cseg + c];
      sv = (kpos <= qpos) ? sv : -3.0e38f;  // causal mask
      vals[c] = sv;
      lmax = fmaxf(lmax, sv);
    }
    __syncthreads();                        // (D) all S reads done (aliases!)
    red[tid] = lmax;
    __syncthreads();                        // (E)
    const float rowmax =
        fmaxf(fmaxf(red[rr * 4 + 0], red[rr * 4 + 1]),
              fmaxf(red[rr * 4 + 2], red[rr * 4 + 3]));
    const float mnew = fmaxf(m_priv, rowmax);
    const float corr = __expf(m_priv - mnew);
    m_priv = mnew;
    __syncthreads();                        // (F) red reads done
    float lsum = 0.0f;
#pragma unroll
    for (int c = 0; c < 16; ++c) {
      const float p = __expf(vals[c] - mnew);
      lsum += p;
      Pt[rr * 64 + cseg + c] = f2bf_raw(p);
    }
    red[tid] = lsum;
    __syncthreads();                        // (G)
    const float ltot = red[rr * 4 + 0] + red[rr * 4 + 1] +
                       red[rr * 4 + 2] + red[rr * 4 + 3];
    l_priv = l_priv * corr + ltot;
    if ((tid & 3) == 0) { cI[rr] = corr; lIs[rr] = l_priv; }
    __syncthreads();                        // (H) cI / Pt ready

    // Rescale O, accumulate O += P @ V. Wave (wid&3)=m-tile, (wid>>2)=64-col half.
    const int om  = wid & 3;
    const int onb = (wid >> 2) << 6;
#pragma unroll
    for (int j = 0; j < 4; ++j)
#pragma unroll
      for (int r = 0; r < 8; ++r)
        oacc[j][r] *= cI[om * 16 + r + rhi];
#pragma unroll
    for (int ks = 0; ks < 2; ++ks) {
      const v16bf ap = fragA16(Pt, om * 16, 64, ks * 32);
#pragma unroll
      for (int j = 0; j < 4; ++j)
        oacc[j] = wmma_bf16(ap, fragB16(Vt, onb + j * 16, 64, ks * 32), oacc[j]);
    }
  }

  // Normalize and store (bf16) into [B*T, H] attention output.
  const int om  = wid & 3;
  const int onb = (wid >> 2) << 6;
#pragma unroll
  for (int j = 0; j < 4; ++j)
#pragma unroll
    for (int r = 0; r < 8; ++r) {
      const int row = om * 16 + r + rhi;
      const float ov = oacc[j][r] / lIs[row];
      obuf[(rowbase + row) * Hc + hoff + onb + j * 16 + nloc] = f2bf_raw(ov);
    }
}

// ---------------------------------------------------------------------------
// Launcher
// ---------------------------------------------------------------------------
extern "C" void kernel_launch(void* const* d_in, const int* in_sizes, int n_in,
                              void* d_out, int out_size, void* d_ws, size_t ws_size,
                              hipStream_t stream) {
  (void)in_sizes; (void)n_in; (void)out_size; (void)ws_size;
  const float* x    = (const float*)d_in[0];
  const float* wq   = (const float*)d_in[1];
  const float* wk   = (const float*)d_in[2];
  const float* wv   = (const float*)d_in[3];
  const float* wo   = (const float*)d_in[4];
  const float* ln1s = (const float*)d_in[5];
  const float* ln1b = (const float*)d_in[6];
  const float* ln2s = (const float*)d_in[7];
  const float* ln2b = (const float*)d_in[8];
  const float* w1   = (const float*)d_in[9];
  const float* b1   = (const float*)d_in[10];
  const float* w2   = (const float*)d_in[11];
  const float* b2   = (const float*)d_in[12];

  unsigned char* ws = (unsigned char*)d_ws;
  size_t off = 0;
  auto carve = [&](size_t bytes) -> void* {
    void* p = ws + off;
    off += (bytes + 255) & ~(size_t)255;
    return p;
  };
  unsigned short* wqT = (unsigned short*)carve((size_t)Hc * Hc * 2);
  unsigned short* wkT = (unsigned short*)carve((size_t)Hc * Hc * 2);
  unsigned short* wvT = (unsigned short*)carve((size_t)Hc * Hc * 2);
  unsigned short* woT = (unsigned short*)carve((size_t)Hc * Hc * 2);
  unsigned short* w1T = (unsigned short*)carve((size_t)Hc * H4c * 2);
  unsigned short* w2T = (unsigned short*)carve((size_t)Hc * H4c * 2);
  unsigned short* h1  = (unsigned short*)carve((size_t)BTc * Hc * 2);
  unsigned short* qB  = (unsigned short*)carve((size_t)BTc * Hc * 2);
  unsigned short* kB  = (unsigned short*)carve((size_t)BTc * Hc * 2);
  unsigned short* vB  = (unsigned short*)carve((size_t)BTc * Hc * 2);
  unsigned short* aB  = (unsigned short*)carve((size_t)BTc * Hc * 2);
  float*          x1  = (float*)        carve((size_t)BTc * Hc * 4);
  unsigned short* h2  = (unsigned short*)carve((size_t)BTc * Hc * 2);
  unsigned short* uB  = (unsigned short*)carve((size_t)BTc * H4c * 2);

  dim3 gtHH(Hc / 64, Hc / 64);
  dim3 gtH14(H4c / 64, Hc / 64);             // w1: K=H rows, N=4H cols
  dim3 gtH41(Hc / 64, H4c / 64);             // w2: K=4H rows, N=H cols
  cast_transpose_kernel<<<gtHH,  TB, 0, stream>>>(wq, wqT, Hc, Hc);
  cast_transpose_kernel<<<gtHH,  TB, 0, stream>>>(wk, wkT, Hc, Hc);
  cast_transpose_kernel<<<gtHH,  TB, 0, stream>>>(wv, wvT, Hc, Hc);
  cast_transpose_kernel<<<gtHH,  TB, 0, stream>>>(wo, woT, Hc, Hc);
  cast_transpose_kernel<<<gtH14, TB, 0, stream>>>(w1, w1T, Hc, H4c);
  cast_transpose_kernel<<<gtH41, TB, 0, stream>>>(w2, w2T, H4c, Hc);

  layernorm_bf16_kernel<<<BTc, TB, 0, stream>>>(x, ln1s, ln1b, h1);

  dim3 gqkv(Hc / 128, BTc / 128);
  gemm_bf16_kernel<EPI_BF16><<<gqkv, TB, 0, stream>>>(h1, wqT, nullptr, nullptr, qB, BTc, Hc, Hc);
  gemm_bf16_kernel<EPI_BF16><<<gqkv, TB, 0, stream>>>(h1, wkT, nullptr, nullptr, kB, BTc, Hc, Hc);
  gemm_bf16_kernel<EPI_BF16><<<gqkv, TB, 0, stream>>>(h1, wvT, nullptr, nullptr, vB, BTc, Hc, Hc);

  dim3 gfa(Tc / 64, NHc, Bc);
  const size_t smemFA = (size_t)(64 * 128 + 64 * 128 + 128 * 64) * 2 + 64 * 64 * 4; // 64 KB
  flash_attn_kernel<<<gfa, TB, smemFA, stream>>>(qB, kB, vB, aB);

  gemm_bf16_kernel<EPI_RES_F32><<<gqkv, TB, 0, stream>>>(aB, woT, nullptr, x, x1, BTc, Hc, Hc);

  layernorm_bf16_kernel<<<BTc, TB, 0, stream>>>(x1, ln2s, ln2b, h2);

  dim3 gffn1(H4c / 128, BTc / 128);
  gemm_bf16_kernel<EPI_BIAS_GELU_BF16><<<gffn1, TB, 0, stream>>>(h2, w1T, b1, nullptr, uB, BTc, H4c, Hc);
  gemm_bf16_kernel<EPI_BIAS_RES_F32><<<gqkv, TB, 0, stream>>>(uB, w2T, b2, x1, d_out, BTc, Hc, H4c);
}
